// GraphMultiHeadAttention_20899310863107
// MI455X (gfx1250) — matrix operand order repaired and checked by reading, not compiled, not run on hardware
//
#include <hip/hip_runtime.h>
#include <hip/hip_bf16.h>
#include <math.h>

// ---------------------------------------------------------------------------
// GraphMultiHeadAttention for MI455X (gfx1250, wave32, WMMA + TDM).
// Memory-bound (~1 GB traffic/call, 536 MB dense attn map) => bf16 WMMA for
// GEMMs (compute is free), TDM (tensor_load_to_lds) to stage weight panels in
// LDS, NT stores for streaming writes, coalesced hw f32 atomics for the
// segment-sum, fused edge-projection+score (edge_emb never materialized).
// ---------------------------------------------------------------------------

constexpr int   Nn    = 4096;
constexpr int   Ee    = 131072;
constexpr int   HIDd  = 256;
constexpr int   Hh    = 8;
constexpr int   Dd    = 32;     // HID / H
constexpr int   EDIMd = 64;
constexpr float SCALEf = 0.17677669529663689f;   // 1/sqrt(32)

constexpr size_t NH = (size_t)Nn * HIDd;         // 1,048,576
constexpr size_t EH = (size_t)Ee * Hh;           // 1,048,576
constexpr int    RED_BLOCKS = 512;               // partial-reduction grid

typedef __attribute__((ext_vector_type(16))) __bf16       v16bf;
typedef __attribute__((ext_vector_type(8)))  float        v8f;
typedef __attribute__((ext_vector_type(4)))  float        f4;
typedef __attribute__((ext_vector_type(4)))  unsigned int u32x4;
typedef __attribute__((ext_vector_type(8)))  int          i32x8;
typedef __attribute__((ext_vector_type(4)))  int          i32x4;

__device__ inline v8f wmma_bf16(v16bf a, v16bf b, v8f c) {
  // D = A(16x32 bf16) * B(32x16 bf16) + C(16x16 f32)
  return __builtin_amdgcn_wmma_f32_16x16x32_bf16(
      /*neg_a=*/false, a, /*neg_b=*/false, b,
      /*c_mod=*/(short)0, c, /*reuse_a=*/false, /*reuse_b=*/false);
}

// ---------------------------------------------------------------------------
// Tensor Data Mover: DMA a 2-D f32 tile (tile_d0 x tile_d1, row stride
// stride0 elements) from global memory into LDS. D# built per CDNA5 ISA
// (group0: count/lds_addr/global_addr/type=2; group1: data_size=4B + dims;
// groups 2/3 zero => up-to-2D tensor). TENSORcnt-tracked.
// Call from ONE wave per block, then s_wait_tensorcnt + __syncthreads().
// Toolchain uses the 6-arg builtin: (u32x4, i32x8, i32x4, i32x4, i32x8, cpol).
// ---------------------------------------------------------------------------
__device__ inline void tdm_load_2d_f32(void* lds_dst, const float* gsrc,
                                       unsigned tile_d0, unsigned tile_d1,
                                       unsigned tensor_d0, unsigned tensor_d1,
                                       unsigned stride0) {
  const unsigned long long ga = (unsigned long long)gsrc;
  const unsigned lds_off = (unsigned)(size_t)lds_dst;   // addr[31:0] = LDS offset
  u32x4 g0;
  g0[0] = 1u;                                       // count=1 (valid user D#)
  g0[1] = lds_off;                                  // lds_addr (bytes)
  g0[2] = (unsigned)(ga & 0xffffffffu);             // global_addr[31:0]
  g0[3] = (unsigned)((ga >> 32) & 0x01ffffffu)      // global_addr[56:32]
        | (2u << 30);                               // type=2 ("image")
  i32x8 g1;
  g1[0] = (int)(2u << 16);                          // data_size=4B, wg_mask=0
  g1[1] = (int)((tensor_d0 & 0xffffu) << 16);       // tensor_dim0[15:0]
  g1[2] = (int)((tensor_d0 >> 16) | ((tensor_d1 & 0xffffu) << 16));
  g1[3] = (int)((tensor_d1 >> 16) | (tile_d0 << 16));   // tile_dim0
  g1[4] = (int)(tile_d1 & 0xffffu);                 // tile_dim1 (tile_dim2=0)
  g1[5] = (int)stride0;                             // tensor_dim0_stride[31:0]
  g1[6] = 0;
  g1[7] = 0;
  const i32x4 z4 = {0, 0, 0, 0};
  const i32x8 z8 = {0, 0, 0, 0, 0, 0, 0, 0};
  __builtin_amdgcn_tensor_load_to_lds(g0, g1, z4, z4, z8, 0);
}

// ---------------------------------------------------------------------------
// GEMM block: 4 waves, each computing a 16x16 tile of a 16x64 output panel.
// A fragments: two contiguous 8-float runs per lane -> 4x global_load_b128.
// B fragments: fed from the LDS weight panel staged by the TDM.
// Wlds layout: [Kdim rows][64 cols] f32.
// ---------------------------------------------------------------------------
__device__ inline void gemm_panel_16x64(const float* __restrict__ A,
                                        const float* __restrict__ Wlds,
                                        const float* __restrict__ bias,
                                        float* __restrict__ C,
                                        int Kdim, int Ncols, int col0) {
  const int wid  = threadIdx.x >> 5;
  const int lane = threadIdx.x & 31;
  const int row0 = blockIdx.y * 16;
  const int arow = row0 + (lane & 15);
  const int wcol = wid * 16 + (lane & 15);   // column inside 64-wide panel
  const int ahi  = (lane >> 4) << 3;         // A: upper half lanes hold k+8/k+24
  const int bhi  = (lane >> 4) << 4;         // unused: B is indexed with ahi, upper half lanes hold k+8/k+24
  v8f acc = {};
  for (int k0 = 0; k0 < Kdim; k0 += 32) {
    const float* ap = A + (size_t)arow * Kdim + k0 + ahi;
    const f4 a0 = *(const f4*)(ap);
    const f4 a1 = *(const f4*)(ap + 4);
    const f4 a2 = *(const f4*)(ap + 16);
    const f4 a3 = *(const f4*)(ap + 20);
    v16bf af, bf2;
#pragma unroll
    for (int i = 0; i < 4; ++i) {
      af[i]      = (__bf16)a0[i];
      af[i + 4]  = (__bf16)a1[i];
      af[i + 8]  = (__bf16)a2[i];
      af[i + 12] = (__bf16)a3[i];
    }
#pragma unroll
    for (int i = 0; i < 16; ++i)
      bf2[i] = (__bf16)Wlds[(size_t)(k0 + ((i >> 3) << 4) + ahi + (i & 7)) * 64 + wcol];
    acc = wmma_bf16(af, bf2, acc);
  }
  const int   mhi = (lane >> 4) << 3;        // C/D: upper lanes = rows +8
  const float bb  = bias[col0 + wcol];
#pragma unroll
  for (int r = 0; r < 8; ++r)
    C[(size_t)(row0 + r + mhi) * Ncols + col0 + wcol] = acc[r] + bb;
}

// Fused Q/K/V projection; blockIdx.z selects weight/bias/output.
__global__ __launch_bounds__(128) void k_gemm_qkv(
    const float* __restrict__ x,
    const float* __restrict__ Wq, const float* __restrict__ bq,
    const float* __restrict__ Wk, const float* __restrict__ bk,
    const float* __restrict__ Wv, const float* __restrict__ bv,
    float* __restrict__ Q, float* __restrict__ K, float* __restrict__ V) {
  __shared__ float Wlds[HIDd * 64];          // 64 KB weight panel
  const float *W, *b;
  float* C;
  if (blockIdx.z == 0)      { W = Wq; b = bq; C = Q; }
  else if (blockIdx.z == 1) { W = Wk; b = bk; C = K; }
  else                      { W = Wv; b = bv; C = V; }
  const int col0 = blockIdx.x * 64;
  if (threadIdx.x < 32) {                    // one wave issues the DMA
    tdm_load_2d_f32(Wlds, W + col0, /*tile*/64, HIDd,
                    /*tensor*/HIDd, HIDd, /*stride0*/HIDd);
    __builtin_amdgcn_s_wait_tensorcnt(0);
  }
  __syncthreads();
  gemm_panel_16x64(x, Wlds, b, C, HIDd, HIDd, col0);
}

// Output projection: out = agg @ Wo + bo
__global__ __launch_bounds__(128) void k_gemm_out(
    const float* __restrict__ agg, const float* __restrict__ Wo,
    const float* __restrict__ bo, float* __restrict__ out) {
  __shared__ float Wlds[HIDd * 64];
  const int col0 = blockIdx.x * 64;
  if (threadIdx.x < 32) {
    tdm_load_2d_f32(Wlds, Wo + col0, 64, HIDd, HIDd, HIDd, HIDd);
    __builtin_amdgcn_s_wait_tensorcnt(0);
  }
  __syncthreads();
  gemm_panel_16x64(agg, Wlds, bo, out, HIDd, HIDd, col0);
}

// ---------------------------------------------------------------------------
// Streaming zero-fill with nontemporal 128b stores (536 MB attn map + agg).
// ---------------------------------------------------------------------------
__global__ __launch_bounds__(256) void k_zero_f4(f4* __restrict__ p, size_t n4) {
  size_t i      = (size_t)blockIdx.x * blockDim.x + threadIdx.x;
  size_t stride = (size_t)gridDim.x * blockDim.x;
  f4 z = {0.f, 0.f, 0.f, 0.f};
  for (; i < n4; i += stride) __builtin_nontemporal_store(z, p + i);
}

// ---------------------------------------------------------------------------
// Scores: 4 waves/block, one 16-edge tile per wave, sharing one TDM-staged
// copy of We (64x256 f32 = 64 KB LDS).
//   scores[e,h] = SCALE*q_i.k_j + sbias[h] + q_i . (edge_attr[e] @ We)[h]
// Edge projection: 2 bf16 WMMAs per 16-col block (K=64), dotted against
// gathered q in WMMA C/D layout, reduced across each 16-lane half.
// ---------------------------------------------------------------------------
__global__ __launch_bounds__(128) void k_scores(
    const float* __restrict__ Q, const float* __restrict__ Kx,
    const long long* __restrict__ ei,         // [2,E] int64 (row, col)
    const float* __restrict__ ea,             // [E, 64]
    const float* __restrict__ We,             // [64, 256]
    const float* __restrict__ sb,             // [8]
    float* __restrict__ scores) {             // [E, 8]
  __shared__ float Wlds[EDIMd * HIDd];        // 64 KB, whole We via TDM
  __shared__ float sbuf[4][16][8];
  __shared__ int   rws[4][16], cls[4][16];
  const int wid  = threadIdx.x >> 5;
  const int lane = threadIdx.x & 31;

  if (threadIdx.x < 32) {
    tdm_load_2d_f32(Wlds, We, /*tile*/HIDd, EDIMd,
                    /*tensor*/HIDd, EDIMd, /*stride0*/HIDd);
    __builtin_amdgcn_s_wait_tensorcnt(0);
  }

  const size_t e0 = ((size_t)blockIdx.x * 4 + wid) * 16;
  if (lane < 16) {
    rws[wid][lane] = (int)ei[e0 + lane];               // destination (row)
    cls[wid][lane] = (int)ei[(size_t)Ee + e0 + lane];  // source (col)
  }
  __syncthreads();

  // --- (1) q.k dots + structural bias: 128 (edge,head) pairs over 32 lanes
  for (int p = lane; p < 128; p += 32) {
    const int m = p >> 3, h = p & 7;
    const float* q = Q  + (size_t)rws[wid][m] * HIDd + h * Dd;
    const float* k = Kx + (size_t)cls[wid][m] * HIDd + h * Dd;
    float d = 0.f;
#pragma unroll 8
    for (int i = 0; i < Dd; ++i) d = fmaf(q[i], k[i], d);
    sbuf[wid][m][h] = d * SCALEf + sb[h];
  }

  // --- (2) edge-feature bias via WMMA (A = edge_attr tile, 16x64 bf16)
  const int am  = lane & 15;
  const int ahi = (lane >> 4) << 3;
  v16bf a0, a1;
#pragma unroll
  for (int i = 0; i < 16; ++i) {
    const int ka = ((i >> 3) << 4) + ahi + (i & 7);
    a0[i] = (__bf16)ea[(e0 + am) * EDIMd + ka];
    a1[i] = (__bf16)ea[(e0 + am) * EDIMd + 32 + ka];
  }
  const int bn  = lane & 15;
  const int bhi = (lane >> 4) << 4;
  const int mlo = (lane >> 4) << 3;
  for (int c = 0; c < 16; ++c) {       // 16 column blocks of 16 (2 per head)
    v16bf b0, b1;
#pragma unroll
    for (int i = 0; i < 16; ++i) {
      b0[i] = (__bf16)Wlds[(size_t)(((i >> 3) << 4) + ahi + (i & 7)) * HIDd + c * 16 + bn];
      b1[i] = (__bf16)Wlds[(size_t)(32 + ((i >> 3) << 4) + ahi + (i & 7)) * HIDd + c * 16 + bn];
    }
    v8f acc = {};
    acc = wmma_bf16(a0, b0, acc);
    acc = wmma_bf16(a1, b1, acc);
    const int h = c >> 1;              // 32 cols per head
    float part[8];
#pragma unroll
    for (int r = 0; r < 8; ++r) {
      const float qv = Q[(size_t)rws[wid][r + mlo] * HIDd + c * 16 + bn];
      part[r] = acc[r] * qv;
    }
#pragma unroll
    for (int mask = 1; mask < 16; mask <<= 1)
#pragma unroll
      for (int r = 0; r < 8; ++r)
        part[r] += __shfl_xor(part[r], mask, 32);
    if (bn == 0) {                     // lane 0: rows 0..7, lane 16: rows 8..15
#pragma unroll
      for (int r = 0; r < 8; ++r) sbuf[wid][r + mlo][h] += part[r];
    }
  }
  __syncthreads();

  for (int p = lane; p < 128; p += 32) {
    const int m = p >> 3, h = p & 7;
    scores[(e0 + m) * Hh + h] = sbuf[wid][m][h];
  }
}

// ---------------------------------------------------------------------------
// Global softmax over all E edges per head. Stride-8 trees keep h = tid&7
// invariant through the reductions.
// ---------------------------------------------------------------------------
__global__ __launch_bounds__(256) void k_max_part(const float* __restrict__ s,
                                                  float* __restrict__ part) {
  __shared__ float red[256];
  const int t = threadIdx.x;
  size_t i      = (size_t)blockIdx.x * 256 + t;
  size_t stride = (size_t)gridDim.x * 256;
  float m = -3.4e38f;
  for (; i < EH; i += stride) m = fmaxf(m, s[i]);
  red[t] = m; __syncthreads();
  for (int st = 128; st >= 8; st >>= 1) {
    if (t < st) red[t] = fmaxf(red[t], red[t + st]);
    __syncthreads();
  }
  if (t < 8) part[blockIdx.x * 8 + t] = red[t];
}

__global__ __launch_bounds__(256) void k_max_fin(const float* __restrict__ part,
                                                 int n, float* __restrict__ fmax) {
  __shared__ float red[256];
  const int t = threadIdx.x;
  float m = -3.4e38f;
  for (int i = t; i < n; i += 256) m = fmaxf(m, part[i]);
  red[t] = m; __syncthreads();
  for (int st = 128; st >= 8; st >>= 1) {
    if (t < st) red[t] = fmaxf(red[t], red[t + st]);
    __syncthreads();
  }
  if (t < 8) fmax[t] = red[t];
}

__global__ __launch_bounds__(256) void k_exp_part(const float* __restrict__ s,
                                                  const float* __restrict__ fmax,
                                                  float* __restrict__ expsc,
                                                  float* __restrict__ part) {
  __shared__ float red[256];
  const int t = threadIdx.x;
  const float mh = fmax[t & 7];
  size_t i      = (size_t)blockIdx.x * 256 + t;
  size_t stride = (size_t)gridDim.x * 256;
  float sum = 0.f;
  for (; i < EH; i += stride) {
    const float p = __expf(s[i] - mh);
    expsc[i] = p;
    sum += p;
  }
  red[t] = sum; __syncthreads();
  for (int st = 128; st >= 8; st >>= 1) {
    if (t < st) red[t] += red[t + st];
    __syncthreads();
  }
  if (t < 8) part[blockIdx.x * 8 + t] = red[t];
}

__global__ __launch_bounds__(256) void k_sum_fin(const float* __restrict__ part,
                                                 int n, float* __restrict__ fsum) {
  __shared__ float red[256];
  const int t = threadIdx.x;
  float sum = 0.f;
  for (int i = t; i < n; i += 256) sum += part[i];
  red[t] = sum; __syncthreads();
  for (int st = 128; st >= 8; st >>= 1) {
    if (t < st) red[t] += red[t + st];
    __syncthreads();
  }
  if (t < 8) fsum[t] = red[t];
}

// ---------------------------------------------------------------------------
// One block per edge; thread t = channel hd in [0,256) (h = hd>>5, d = hd&31).
// - Coalesced read of V[col] (1 KB/edge) and coalesced global_atomic_add_f32
//   bursts into agg[row] (4 MB, L2-resident).
// - Threads 0..7 scatter the 8 per-head probs into the dense [N,N,H] map
//   with NT stores (write-once lines).
// Edge indices are block-uniform -> scalar loads.
// ---------------------------------------------------------------------------
__global__ __launch_bounds__(256) void k_scatter(
    const float* __restrict__ expsc, const float* __restrict__ fsum,
    const long long* __restrict__ ei, const float* __restrict__ V,
    float* __restrict__ agg, float* __restrict__ attn) {
  const size_t e  = blockIdx.x;
  const int    hd = threadIdx.x;
  const int    h  = hd >> 5;
  const int    r  = (int)ei[e];
  const int    c  = (int)ei[(size_t)Ee + e];
  const float  p  = expsc[e * Hh + h] / fsum[h];
  if (hd < Hh) {
    const float ph = expsc[e * Hh + hd] / fsum[hd];
    __builtin_nontemporal_store(ph, &attn[((size_t)r * Nn + c) * Hh + hd]);
  }
  const float v = V[(size_t)c * HIDd + hd];
  unsafeAtomicAdd(&agg[(size_t)r * HIDd + hd], p * v);  // global_atomic_add_f32
}

// ---------------------------------------------------------------------------
extern "C" void kernel_launch(void* const* d_in, const int* in_sizes, int n_in,
                              void* d_out, int out_size, void* d_ws, size_t ws_size,
                              hipStream_t stream) {
  const float*     x  = (const float*)d_in[0];
  const long long* ei = (const long long*)d_in[1];   // int64 [2,E]
  const float*     ea = (const float*)d_in[2];
  const float*     Wq = (const float*)d_in[3];
  const float*     bq = (const float*)d_in[4];
  const float*     Wk = (const float*)d_in[5];
  const float*     bk = (const float*)d_in[6];
  const float*     Wv = (const float*)d_in[7];
  const float*     bv = (const float*)d_in[8];
  const float*     We = (const float*)d_in[9];
  const float*     Wo = (const float*)d_in[10];
  const float*     bo = (const float*)d_in[11];
  const float*     sb = (const float*)d_in[12];

  float* out  = (float*)d_out;                 // [N, HID]
  float* attn = out + NH;                      // [N, N, H] = 536 MB

  // Workspace (~25 MB; edge_emb deliberately NOT materialized).
  float* Q      = (float*)d_ws;                // [N, HID]
  float* K      = Q + NH;                      // [N, HID]
  float* V      = K + NH;                      // [N, HID]
  float* scores = V + NH;                      // [E, H]
  float* expsc  = scores + EH;                 // [E, H]
  float* agg    = expsc + EH;                  // [N, HID]
  float* part   = agg + NH;                    // [RED_BLOCKS * 8]
  float* fmax   = part + RED_BLOCKS * 8;       // [8]
  float* fsum   = fmax + 8;                    // [8]

  const dim3 b128(128), b256(256);

  // 1) Q/K/V projections: 3 x (4096x256x256) bf16 WMMA GEMMs, TDM-fed.
  k_gemm_qkv<<<dim3(HIDd / 64, Nn / 16, 3), b128, 0, stream>>>(
      x, Wq, bq, Wk, bk, Wv, bv, Q, K, V);

  // 2) Zero the dense attention map (streaming NT stores) and agg.
  k_zero_f4<<<4096, b256, 0, stream>>>((f4*)attn, ((size_t)Nn * Nn * Hh) / 4);
  k_zero_f4<<<64,   b256, 0, stream>>>((f4*)agg, NH / 4);

  // 3) Fused scores: q.k dot + sbias + WMMA edge-projection dot (TDM-fed We).
  k_scores<<<Ee / 64, b128, 0, stream>>>(Q, K, ei, ea, We, sb, scores);

  // 4) Global per-head softmax (max -> exp/sum -> normalize-in-scatter).
  k_max_part<<<RED_BLOCKS, b256, 0, stream>>>(scores, part);
  k_max_fin<<<1, b256, 0, stream>>>(part, RED_BLOCKS * 8, fmax);
  k_exp_part<<<RED_BLOCKS, b256, 0, stream>>>(scores, fmax, expsc, part);
  k_sum_fin<<<1, b256, 0, stream>>>(part, RED_BLOCKS * 8, fsum);

  // 5) Scatter probs into attn map + coalesced atomic segment-sum into agg.
  k_scatter<<<Ee, b256, 0, stream>>>(expsc, fsum, ei, V, agg, attn);

  // 6) Output projection: out = agg @ Wo + bo.
  k_gemm_out<<<dim3(HIDd / 64, Nn / 16, 1), b128, 0, stream>>>(agg, Wo, bo, out);
}
